// LoCS_7215545057967
// MI455X (gfx1250) — matrix operand order repaired and checked by reading, not compiled, not run on hardware
//
#include <hip/hip_runtime.h>
#include <hip/hip_bf16.h>
#include <stdint.h>

// ---------------------------------------------------------------------------
// LoCS GNN forward for gfx1250 (MI455X).
//   - All H=64 GEMMs on v_wmma_f32_16x16x32_f16 (f16 in, f32 accumulate).
//   - One wave owns 16 rows x ALL column tiles (A fragment loaded once).
//   - Edge message MLP (m1 -> silu -> m2 -> silu) fused in one kernel via a
//     per-wave LDS staging tile; scatter-add fused into the m2 epilogue.
//   - Edge concat [x_send | x_recv | m_prev] fused into the A-frag gather.
//   - W1/W2 staged block-wide into LDS with global_load_async_to_lds_b128
//     (ASYNCcnt / s_wait_asynccnt), B fragments then come from ds_load.
// ---------------------------------------------------------------------------

typedef _Float16 v16h __attribute__((ext_vector_type(16)));
typedef float    v8f  __attribute__((ext_vector_type(8)));

union AFrag { v16h v; uint32_t u[8]; };

#define SILU(y) ((y) / (1.0f + __expf(-(y))))

// ---------------------------------------------------------------------------
// Weight convert: f32 [rows, kin] -> f16 [rows, kout] (zero-padded K)
// ---------------------------------------------------------------------------
__global__ void k_convert_w(const float* __restrict__ src, _Float16* __restrict__ dst,
                            int rows, int kin, int kout) {
    int i = blockIdx.x * blockDim.x + threadIdx.x;
    if (i >= rows * kout) return;
    int r = i / kout, k = i - r * kout;
    dst[i] = (k < kin) ? (_Float16)src[r * kin + k] : (_Float16)0.0f;
}

// ---------------------------------------------------------------------------
// Per-node localizer: rotation (c,s) from vel, canonical vel.
// ---------------------------------------------------------------------------
__global__ void k_node_pre(const float* __restrict__ vel, float* __restrict__ cs,
                           float* __restrict__ cv, int N) {
    int n = blockIdx.x * blockDim.x + threadIdx.x;
    if (n >= N) return;
    float vx = vel[2 * n], vy = vel[2 * n + 1];
    float th = atan2f(vy, vx);
    float c = cosf(th), s = sinf(th);
    cs[2 * n] = c; cs[2 * n + 1] = s;
    cv[2 * n]     = c * vx + s * vy;   // Rinv . vel
    cv[2 * n + 1] = -s * vx + c * vy;
}

// ---------------------------------------------------------------------------
// Per-edge 13-dim localized features, padded to K=32 f16.
// ---------------------------------------------------------------------------
__global__ void k_edge_feat(const float* __restrict__ x, const float* __restrict__ vel,
                            const int* __restrict__ send, const int* __restrict__ recv,
                            const float* __restrict__ eao, const float* __restrict__ cs,
                            const float* __restrict__ cv, _Float16* __restrict__ ea, int E) {
    int e = blockIdx.x * blockDim.x + threadIdx.x;
    if (e >= E) return;
    int s = send[e], r = recv[e];
    float ci = cs[2 * r], si = cs[2 * r + 1];
    float cj = cs[2 * s], sj = cs[2 * s + 1];
    float dx = x[2 * s] - x[2 * r], dy = x[2 * s + 1] - x[2 * r + 1];
    float rrx = ci * dx + si * dy;
    float rry = -si * dx + ci * dy;
    float euler = atan2f(ci * sj - si * cj, ci * cj + si * sj);
    float dist = sqrtf(dx * dx + dy * dy);
    float sph = atan2f(rry, rrx);
    float vjx = vel[2 * s], vjy = vel[2 * s + 1];
    float rvx = ci * vjx + si * vjy;
    float rvy = -si * vjx + ci * vjy;
    float f[13] = { rrx, rry, euler, dist, sph, rvx, rvy,
                    0.0f, 0.0f, cv[2 * r], cv[2 * r + 1],
                    eao[2 * e], eao[2 * e + 1] };
    _Float16* dst = ea + (size_t)e * 32;
    #pragma unroll
    for (int j = 0; j < 13; ++j) dst[j] = (_Float16)f[j];
    #pragma unroll
    for (int j = 13; j < 32; ++j) dst[j] = (_Float16)0.0f;
}

// ---------------------------------------------------------------------------
// Generic WMMA GEMM: Y[M, CT*16] = act( X[M,K] @ W[CT*16,K]^T + b )
// One wave: 16 rows x all CT column tiles (A fragment loaded once per K tile).
// Optional f32 residual accumulate:  acc += Y; f16 out = acc.
// ---------------------------------------------------------------------------
template <int CT>
__global__ __launch_bounds__(256) void k_gemm(const _Float16* __restrict__ X,
                                              const _Float16* __restrict__ W,
                                              const float* __restrict__ bias,
                                              _Float16* __restrict__ Yh,
                                              float* __restrict__ Yacc,
                                              int M, int K, int act) {
    int wave = (blockIdx.x * blockDim.x + threadIdx.x) >> 5;
    if (wave >= (M >> 4)) return;
    int lane = threadIdx.x & 31;
    int mla = lane & 15, hi = lane >> 4;
    int row0 = wave << 4;
    const int Nout = CT * 16;

    AFrag a, b;
    v8f acc[CT];
    #pragma unroll
    for (int ct = 0; ct < CT; ++ct) acc[ct] = (v8f){0.f,0.f,0.f,0.f,0.f,0.f,0.f,0.f};

    const _Float16* xrow  = X + (size_t)(row0 + mla) * K;
    const _Float16* wbase = W + (size_t)mla * K;          // column n == mla

    for (int kb = 0; kb < K; kb += 32) {
        #pragma unroll
        for (int v = 0; v < 8; ++v) {
            int ka = kb + ((v >> 2) << 4) + (hi << 3) + ((v & 3) << 1);
            a.u[v] = *(const uint32_t*)(xrow + ka);
        }
        #pragma unroll
        for (int ct = 0; ct < CT; ++ct) {
            const _Float16* wrow = wbase + (size_t)ct * 16 * K;
            #pragma unroll
            for (int v = 0; v < 8; ++v) {
                int kbv = kb + ((v >> 2) << 4) + (hi << 3) + ((v & 3) << 1);
                b.u[v] = *(const uint32_t*)(wrow + kbv);
            }
            acc[ct] = __builtin_amdgcn_wmma_f32_16x16x32_f16(false, a.v, false, b.v,
                                                             (short)0, acc[ct], false, false);
        }
    }
    #pragma unroll
    for (int ct = 0; ct < CT; ++ct) {
        int colg = ct * 16 + mla;
        float bia = bias ? bias[colg] : 0.0f;
        #pragma unroll
        for (int i = 0; i < 8; ++i) {
            int row = row0 + (hi << 3) + i;
            float y = acc[ct][i] + bia;
            if (act) y = SILU(y);
            size_t idx = (size_t)row * Nout + colg;
            if (Yacc) { float t = Yacc[idx] + y; Yacc[idx] = t; y = t; }
            if (Yh) Yh[idx] = (_Float16)y;
        }
    }
}

// ---------------------------------------------------------------------------
// Fused edge message MLP:  m = silu( W2 . silu( W1 . in + b1 ) + b2 )
//   MODE 0 (layer 1):   in = ea[e]                      (K1 = 32)
//   MODE 1 (layers 2+): in = [x[send]|x[recv]|m_prev]   (K1 = 192, gathered)
// W1/W2 staged block-wide into LDS via global_load_async_to_lds_b128.
// Phase 1 result staged in a 2KB/wave LDS tile (D-frag -> A-frag relayout),
// phase 2 GEMM reads A from LDS, epilogue writes m and fuses the scatter-add
// (global_atomic_add_f32 into aggr[recv]).
// ---------------------------------------------------------------------------
template <int MODE>
__global__ __launch_bounds__(256) void k_edge_mlp(const _Float16* __restrict__ src,
                                                  const _Float16* __restrict__ mprev,
                                                  const int* __restrict__ send,
                                                  const int* __restrict__ recv,
                                                  const _Float16* __restrict__ W1,
                                                  const float* __restrict__ b1,
                                                  const _Float16* __restrict__ W2,
                                                  const float* __restrict__ b2,
                                                  _Float16* __restrict__ mout,
                                                  float* __restrict__ aggr, int E) {
    constexpr int K1 = MODE ? 192 : 32;
    __shared__ __align__(16) _Float16 tile[8][16][64];    // 16 KB: inter-phase staging
    __shared__ __align__(16) _Float16 sW1[64 * K1];       // 24 KB (MODE1) / 4 KB
    __shared__ __align__(16) _Float16 sW2[64 * 64];       // 8 KB

    // ---- block-wide async stage of W1 and W2 into LDS (16B granules) ----
    {
        const int c1 = (64 * K1 * 2) >> 4;                // W1 16B chunks
        const int ctot = c1 + ((64 * 64 * 2) >> 4);       // + W2 16B chunks
        for (int c = threadIdx.x; c < ctot; c += 256) {
            uint32_t lds;
            uint64_t ga;
            if (c < c1) {
                lds = (uint32_t)(uintptr_t)(&sW1[0]) + (c << 4);
                ga  = (uint64_t)(uintptr_t)W1 + ((uint64_t)c << 4);
            } else {
                int cc = c - c1;
                lds = (uint32_t)(uintptr_t)(&sW2[0]) + (cc << 4);
                ga  = (uint64_t)(uintptr_t)W2 + ((uint64_t)cc << 4);
            }
            asm volatile("global_load_async_to_lds_b128 %0, %1, off"
                         :: "v"(lds), "v"(ga) : "memory");
        }
        asm volatile("s_wait_asynccnt 0x0" ::: "memory");
    }
    __syncthreads();

    int wid  = threadIdx.x >> 5;
    int wave = (blockIdx.x * blockDim.x + threadIdx.x) >> 5;
    int lane = threadIdx.x & 31;
    int mla = lane & 15, hi = lane >> 4;
    bool active = wave < (E >> 4);
    int row0 = wave << 4;

    AFrag a, b;
    if (active) {
        // ---- phase 1: hid = silu(W1 . in + b1), all 4 column tiles ----
        v8f acc[4];
        #pragma unroll
        for (int ct = 0; ct < 4; ++ct) acc[ct] = (v8f){0.f,0.f,0.f,0.f,0.f,0.f,0.f,0.f};

        int row = row0 + mla;
        const _Float16 *seg0, *seg1, *seg2;
        if (MODE) {
            seg0 = src   + ((size_t)send[row] << 6);
            seg1 = src   + ((size_t)recv[row] << 6);
            seg2 = mprev + ((size_t)row << 6);
        } else {
            seg0 = src + ((size_t)row << 5);
            seg1 = seg2 = seg0;
        }

        #pragma unroll
        for (int kt = 0; kt < (MODE ? 6 : 1); ++kt) {
            int kb = kt << 5;
            const _Float16* base =
                (!MODE)      ? seg0
                : (kb < 64)  ? (seg0 + kb)
                : (kb < 128) ? (seg1 + (kb - 64))
                             : (seg2 + (kb - 128));
            #pragma unroll
            for (int v = 0; v < 8; ++v) {
                int ka = ((v >> 2) << 4) + (hi << 3) + ((v & 3) << 1);
                a.u[v] = *(const uint32_t*)(base + ka);
            }
            #pragma unroll
            for (int ct = 0; ct < 4; ++ct) {
                #pragma unroll
                for (int v = 0; v < 8; ++v) {
                    int kbv = kb + ((v >> 2) << 4) + (hi << 3) + ((v & 3) << 1);
                    b.u[v] = *(const uint32_t*)&sW1[(ct * 16 + mla) * K1 + kbv];
                }
                acc[ct] = __builtin_amdgcn_wmma_f32_16x16x32_f16(
                    false, a.v, false, b.v, (short)0, acc[ct], false, false);
            }
        }
        // silu + stage into LDS (D layout: row = hi*8+i, col = ct*16+mla)
        #pragma unroll
        for (int ct = 0; ct < 4; ++ct) {
            int colg = ct * 16 + mla;
            float bia = b1[colg];
            #pragma unroll
            for (int i = 0; i < 8; ++i) {
                float y = SILU(acc[ct][i] + bia);
                tile[wid][(hi << 3) + i][colg] = (_Float16)y;
            }
        }
    }
    __syncthreads();
    if (active) {
        // ---- phase 2: m = silu(W2 . hid + b2), A fragments from LDS ----
        v8f acc[4];
        #pragma unroll
        for (int ct = 0; ct < 4; ++ct) acc[ct] = (v8f){0.f,0.f,0.f,0.f,0.f,0.f,0.f,0.f};

        #pragma unroll
        for (int kt = 0; kt < 2; ++kt) {
            int kb = kt << 5;
            #pragma unroll
            for (int v = 0; v < 8; ++v) {
                int ka = kb + ((v >> 2) << 4) + (hi << 3) + ((v & 3) << 1);
                a.u[v] = *(const uint32_t*)&tile[wid][mla][ka];
            }
            #pragma unroll
            for (int ct = 0; ct < 4; ++ct) {
                #pragma unroll
                for (int v = 0; v < 8; ++v) {
                    int kbv = kb + ((v >> 2) << 4) + (hi << 3) + ((v & 3) << 1);
                    b.u[v] = *(const uint32_t*)&sW2[(ct * 16 + mla) * 64 + kbv];
                }
                acc[ct] = __builtin_amdgcn_wmma_f32_16x16x32_f16(
                    false, a.v, false, b.v, (short)0, acc[ct], false, false);
            }
        }
        // epilogue: write m (f16) + fused scatter-add into aggr[recv]
        #pragma unroll
        for (int ct = 0; ct < 4; ++ct) {
            int colg = ct * 16 + mla;
            float bia = b2[colg];
            #pragma unroll
            for (int i = 0; i < 8; ++i) {
                int row = row0 + (hi << 3) + i;
                float y = SILU(acc[ct][i] + bia);
                mout[((size_t)row << 6) + colg] = (_Float16)y;
                atomicAdd(aggr + ((size_t)recv[row] << 6) + colg, y);
            }
        }
    }
}

// ---------------------------------------------------------------------------
// scatter_mean support (cnt depends only on recv -> computed once)
// ---------------------------------------------------------------------------
__global__ void k_cnt(const int* __restrict__ recv, float* __restrict__ cnt, int E) {
    int e = blockIdx.x * blockDim.x + threadIdx.x;
    if (e >= E) return;
    atomicAdd(cnt + recv[e], 1.0f);
}
__global__ void k_inv(const float* __restrict__ cnt, float* __restrict__ inv, int N) {
    int n = blockIdx.x * blockDim.x + threadIdx.x;
    if (n >= N) return;
    inv[n] = 1.0f / fmaxf(cnt[n], 1.0f);
}

// Layer-1 combine: x = res(rel_feat) + aggr/cnt   (rel_feat = [0,0,cv])
__global__ void k_combine1(const float* __restrict__ aggr, const float* __restrict__ inv,
                           const float* __restrict__ cv, const float* __restrict__ resw,
                           const float* __restrict__ resb, float* __restrict__ xf32,
                           _Float16* __restrict__ xf16, int N) {
    int i = blockIdx.x * blockDim.x + threadIdx.x;
    if (i >= (N << 6)) return;
    int n = i >> 6, col = i & 63;
    float xr = resb[col] + resw[col * 4 + 2] * cv[2 * n] + resw[col * 4 + 3] * cv[2 * n + 1];
    float v = xr + aggr[i] * inv[n];
    xf32[i] = v;
    xf16[i] = (_Float16)v;
}
// Layers 2..4 combine: x += aggr/cnt
__global__ void k_combineL(const float* __restrict__ aggr, const float* __restrict__ inv,
                           float* __restrict__ xf32, _Float16* __restrict__ xf16, int N) {
    int i = blockIdx.x * blockDim.x + threadIdx.x;
    if (i >= (N << 6)) return;
    int n = i >> 6;
    float v = xf32[i] + aggr[i] * inv[n];
    xf32[i] = v;
    xf16[i] = (_Float16)v;
}

// ---------------------------------------------------------------------------
// Head: pred = W3(2x64) . t2 + b3 ; rotate back by R ; out = x + pred
// ---------------------------------------------------------------------------
__global__ void k_final(const _Float16* __restrict__ t2, const float* __restrict__ w3,
                        const float* __restrict__ b3, const float* __restrict__ cs,
                        const float* __restrict__ xin, float* __restrict__ out, int N) {
    int n = blockIdx.x * blockDim.x + threadIdx.x;
    if (n >= N) return;
    const _Float16* a = t2 + ((size_t)n << 6);
    float p0 = b3[0], p1 = b3[1];
    #pragma unroll
    for (int k = 0; k < 64; ++k) {
        float av = (float)a[k];
        p0 = fmaf(w3[k], av, p0);
        p1 = fmaf(w3[64 + k], av, p1);
    }
    float c = cs[2 * n], s = cs[2 * n + 1];
    out[2 * n]     = xin[2 * n]     + c * p0 - s * p1;
    out[2 * n + 1] = xin[2 * n + 1] + s * p0 + c * p1;
}

// ===========================================================================
// Host side
// ===========================================================================
struct LinP { const float* b; const float* w; int bn; int wn; };

static LinP pair_at(void* const* d_in, const int* sz, int i) {
    LinP L;
    if (sz[i] <= sz[i + 1]) { L.b = (const float*)d_in[i];     L.bn = sz[i];
                              L.w = (const float*)d_in[i + 1]; L.wn = sz[i + 1]; }
    else                    { L.w = (const float*)d_in[i];     L.wn = sz[i];
                              L.b = (const float*)d_in[i + 1]; L.bn = sz[i + 1]; }
    return L;
}

static inline int ceil_div(long long a, int b) { return (int)((a + b - 1) / b); }

extern "C" void kernel_launch(void* const* d_in, const int* in_sizes, int n_in,
                              void* d_out, int out_size, void* d_ws, size_t ws_size,
                              hipStream_t stream) {
    const int N = in_sizes[1] / 2;     // x is [N,2]
    const int E = in_sizes[3];         // send_edges

    const float* x_in  = (const float*)d_in[1];
    const float* vel   = (const float*)d_in[2];
    const int*   send  = (const int*)d_in[3];
    const int*   recv  = (const int*)d_in[4];
    const float* eao   = (const float*)d_in[5];

    // ---- resolve params by size signature (robust to leaf ordering) ----
    LinP m1[4], m2[4], u1[4], u2[4], res, o1, o2, o3;
    int idx = 6;
    for (int p = 0; p < 5; ++p) {            // l1: m1,m2,res,u1,u2 in some order
        LinP L = pair_at(d_in, in_sizes, idx); idx += 2;
        if      (L.wn == 64 * 13)  m1[0] = L;
        else if (L.wn == 64 * 64)  m2[0] = L;
        else if (L.wn == 64 * 4)   res   = L;
        else if (L.bn == 128)      u1[0] = L;
        else                       u2[0] = L;
    }
    for (int l = 1; l < 4; ++l) {
        for (int p = 0; p < 4; ++p) {
            LinP L = pair_at(d_in, in_sizes, idx); idx += 2;
            if      (L.wn == 64 * 192) m1[l] = L;
            else if (L.wn == 64 * 64)  m2[l] = L;
            else if (L.bn == 128)      u1[l] = L;
            else                       u2[l] = L;
        }
    }
    {
        int seen4096 = 0;
        for (int p = 0; p < 3; ++p) {
            LinP L = pair_at(d_in, in_sizes, idx); idx += 2;
            if (L.wn == 2 * 64) o3 = L;
            else { if (seen4096++ == 0) o1 = L; else o2 = L; }
        }
    }

    // ---- workspace layout ----
    char* wsb = (char*)d_ws;
    size_t off = 0;
    auto alloc = [&](size_t bytes) -> void* {
        off = (off + 255) & ~(size_t)255;
        void* p = wsb + off;
        off += bytes;
        return p;
    };
    float*     cs    = (float*)alloc((size_t)N * 2 * 4);
    float*     cv    = (float*)alloc((size_t)N * 2 * 4);
    float*     cnt   = (float*)alloc((size_t)N * 4);
    float*     inv   = (float*)alloc((size_t)N * 4);
    _Float16*  ea    = (_Float16*)alloc((size_t)E * 32 * 2);
    float*     xf32  = (float*)alloc((size_t)N * 64 * 4);
    _Float16*  xf16  = (_Float16*)alloc((size_t)N * 64 * 2);
    _Float16*  mbuf  = (_Float16*)alloc((size_t)E * 64 * 2);
    float*     aggr  = (float*)alloc((size_t)N * 64 * 4);
    _Float16*  h1    = (_Float16*)alloc((size_t)N * 128 * 2);
    _Float16*  t1    = (_Float16*)alloc((size_t)N * 64 * 2);
    _Float16*  t2    = (_Float16*)alloc((size_t)N * 64 * 2);
    _Float16*  wm1[4], *wm2[4], *wu1[4], *wu2[4], *wo1, *wo2;
    wm1[0] = (_Float16*)alloc(64 * 32 * 2);
    for (int l = 1; l < 4; ++l) wm1[l] = (_Float16*)alloc(64 * 192 * 2);
    for (int l = 0; l < 4; ++l) wm2[l] = (_Float16*)alloc(64 * 64 * 2);
    for (int l = 0; l < 4; ++l) wu1[l] = (_Float16*)alloc(128 * 64 * 2);
    for (int l = 0; l < 4; ++l) wu2[l] = (_Float16*)alloc(64 * 128 * 2);
    wo1 = (_Float16*)alloc(64 * 64 * 2);
    wo2 = (_Float16*)alloc(64 * 64 * 2);
    (void)ws_size;

    // ---- weight conversion to f16 ----
    auto conv = [&](const float* src, _Float16* dst, int rows, int kin, int kout) {
        int tot = rows * kout;
        k_convert_w<<<ceil_div(tot, 256), 256, 0, stream>>>(src, dst, rows, kin, kout);
    };
    conv(m1[0].w, wm1[0], 64, 13, 32);
    for (int l = 1; l < 4; ++l) conv(m1[l].w, wm1[l], 64, 192, 192);
    for (int l = 0; l < 4; ++l) conv(m2[l].w, wm2[l], 64, 64, 64);
    for (int l = 0; l < 4; ++l) conv(u1[l].w, wu1[l], 128, 64, 64);
    for (int l = 0; l < 4; ++l) conv(u2[l].w, wu2[l], 64, 128, 128);
    conv(o1.w, wo1, 64, 64, 64);
    conv(o2.w, wo2, 64, 64, 64);

    // ---- localizer ----
    k_node_pre<<<ceil_div(N, 256), 256, 0, stream>>>(vel, cs, cv, N);
    hipMemsetAsync(cnt, 0, (size_t)N * 4, stream);
    k_cnt<<<ceil_div(E, 256), 256, 0, stream>>>(recv, cnt, E);
    k_inv<<<ceil_div(N, 256), 256, 0, stream>>>(cnt, inv, N);
    k_edge_feat<<<ceil_div(E, 256), 256, 0, stream>>>(x_in, vel, send, recv, eao, cs, cv, ea, E);

    auto gemm4 = [&](const _Float16* X, const _Float16* W, const float* b,
                     _Float16* Yh, float* Yacc, int M, int K, int act) {
        int blocks = ceil_div((long long)(M / 16) * 32, 256);
        k_gemm<4><<<blocks, 256, 0, stream>>>(X, W, b, Yh, Yacc, M, K, act);
    };
    auto gemm8 = [&](const _Float16* X, const _Float16* W, const float* b,
                     _Float16* Yh, float* Yacc, int M, int K, int act) {
        int blocks = ceil_div((long long)(M / 16) * 32, 256);
        k_gemm<8><<<blocks, 256, 0, stream>>>(X, W, b, Yh, Yacc, M, K, act);
    };

    // ---- GNN layers ----
    int eblocks = ceil_div((long long)(E / 16) * 32, 256);
    for (int l = 0; l < 4; ++l) {
        hipMemsetAsync(aggr, 0, (size_t)N * 64 * 4, stream);
        // fused message MLP (m1 -> silu -> m2 -> silu) + scatter-add
        if (l == 0)
            k_edge_mlp<0><<<eblocks, 256, 0, stream>>>(
                ea, nullptr, nullptr, recv, wm1[0], m1[0].b, wm2[0], m2[0].b, mbuf, aggr, E);
        else
            k_edge_mlp<1><<<eblocks, 256, 0, stream>>>(
                xf16, mbuf, send, recv, wm1[l], m1[l].b, wm2[l], m2[l].b, mbuf, aggr, E);
        // combine (mean + residual path)
        if (l == 0)
            k_combine1<<<ceil_div((long long)N * 64, 256), 256, 0, stream>>>(
                aggr, inv, cv, res.w, res.b, xf32, xf16, N);
        else
            k_combineL<<<ceil_div((long long)N * 64, 256), 256, 0, stream>>>(
                aggr, inv, xf32, xf16, N);
        // node update MLP (residual accumulated into xf32, xf16 refreshed)
        gemm8(xf16, wu1[l], u1[l].b, h1, nullptr, N, 64, 1);
        gemm4(h1, wu2[l], u2[l].b, xf16, xf32, N, 128, 0);
    }

    // ---- output head ----
    gemm4(xf16, wo1, o1.b, t1, nullptr, N, 64, 1);
    gemm4(t1, wo2, o2.b, t2, nullptr, N, 64, 1);
    k_final<<<ceil_div(N, 256), 256, 0, stream>>>(t2, o3.w, o3.b, cs, x_in, (float*)d_out, N);
}
